// Decoder_42210938585769
// MI455X (gfx1250) — matrix operand order repaired and checked by reading, not compiled, not run on hardware
//
#include <hip/hip_runtime.h>

#define B_   32
#define LS_  64
#define LT_  64
#define D_   512
#define NWD_ 32000
#define NH_  8
#define DH_  64
#define NGRP_CLS (NWD_ / 64)   // 500

typedef __attribute__((ext_vector_type(16))) __bf16 v16bf;
typedef __attribute__((ext_vector_type(8)))  float  v8f;

union FragB16 {
    unsigned u[8];
    uint4    q[2];
    v16bf    v;
};

__device__ __forceinline__ unsigned short f2bf(float f) {
    unsigned u = __builtin_bit_cast(unsigned, f);
    u = (u + 0x7fffu + ((u >> 16) & 1u)) >> 16;
    return (unsigned short)u;
}
__device__ __forceinline__ v8f wmma_bf16(const FragB16& a, const FragB16& b, v8f c) {
    return __builtin_amdgcn_wmma_f32_16x16x32_bf16(false, a.v, false, b.v, (short)0, c, false, false);
}
__device__ __forceinline__ float sigm(float x) { return 1.0f / (1.0f + __expf(-x)); }

// ---------------- device-wide barrier (persistent LSTM scan) ----------------
__device__ __forceinline__ void grid_barrier(unsigned* cnt, unsigned* gen, unsigned nblocks) {
    __syncthreads();
    if (threadIdx.x == 0) {
        __threadfence();
        unsigned g = __hip_atomic_load(gen, __ATOMIC_ACQUIRE, __HIP_MEMORY_SCOPE_AGENT);
        unsigned prev = __hip_atomic_fetch_add(cnt, 1u, __ATOMIC_ACQ_REL, __HIP_MEMORY_SCOPE_AGENT);
        if (prev == nblocks - 1u) {
            __hip_atomic_store(cnt, 0u, __ATOMIC_RELAXED, __HIP_MEMORY_SCOPE_AGENT);
            __hip_atomic_fetch_add(gen, 1u, __ATOMIC_ACQ_REL, __HIP_MEMORY_SCOPE_AGENT);
        } else {
            while (__hip_atomic_load(gen, __ATOMIC_ACQUIRE, __HIP_MEMORY_SCOPE_AGENT) == g)
                __builtin_amdgcn_s_sleep(1);
        }
        __threadfence();
    }
    __syncthreads();
}

__global__ void init_bar_kernel(unsigned* bar) {
    if (threadIdx.x < 2) bar[threadIdx.x] = 0u;
}

// -------- pack f32 weight (K x N row-major) into WMMA B-fragment order ------
__global__ void pack_w_kernel(const float* __restrict__ W, unsigned short* __restrict__ P,
                              int K, int N) {
    int tid = blockIdx.x * blockDim.x + threadIdx.x;
    int KC = K >> 5, NT = N >> 4;
    int total = KC * NT * 32;
    if (tid >= total) return;
    int lane = tid & 31;
    int kc   = (tid >> 5) % KC;
    int nt   = (tid >> 5) / KC;
    int n     = nt * 16 + (lane & 15);
    int kbase = kc * 32 + ((lane >> 4) << 3);
    unsigned short* dst = P + (size_t)tid * 16;
#pragma unroll
    for (int e = 0; e < 16; ++e)
        dst[e] = f2bf(W[(size_t)(kbase + e + ((e >> 3) << 3)) * N + n]);
}

// ----------------------- f32 -> bf16 elementwise ----------------------------
__global__ void cvt_bf16_kernel(const float* __restrict__ X, unsigned short* __restrict__ Y, int n) {
    int i = blockIdx.x * blockDim.x + threadIdx.x;
    if (i < n) Y[i] = f2bf(X[i]);
}

// --------------------- embedding gather (bf16 out) --------------------------
__global__ void embed_kernel(const int* __restrict__ ids, const float* __restrict__ wemb,
                             unsigned short* __restrict__ x0) {
    int idx = blockIdx.x * blockDim.x + threadIdx.x;
    if (idx >= B_ * LT_ * D_) return;
    int d = idx & (D_ - 1);
    int row = idx >> 9;
    x0[idx] = f2bf(wemb[(size_t)ids[row] * D_ + d]);
}

// ---- generic WMMA GEMM: 32(M) x 64(N) per wave, double-buffered, K=512 -----
__global__ __launch_bounds__(256) void gemm_ws_kernel(
    const unsigned short* __restrict__ A, int lda,
    const unsigned short* __restrict__ Bp, int kcOff, int KCtot,
    float* __restrict__ C, int ldc, const float* __restrict__ bias,
    unsigned short* __restrict__ Cbf,
    int M, int N, int accFlag) {
    int gwave = (blockIdx.x * blockDim.x + threadIdx.x) >> 5;
    int lane  = threadIdx.x & 31;
    int ngroups = N >> 6;
    int total   = (M >> 5) * ngroups;
    if (gwave >= total) return;
    int mt = gwave / ngroups, ng = gwave % ngroups;
    int lh = lane >> 4;

    const unsigned short* Abase =
        A + (size_t)(mt * 32 + (lane & 15)) * lda + lh * 8;
    const unsigned short* Bbase =
        Bp + ((size_t)(ng * 4 * KCtot + kcOff) * 32 + lane) * 16;

    v8f acc[2][4];
#pragma unroll
    for (int mi = 0; mi < 2; ++mi)
#pragma unroll
        for (int t = 0; t < 4; ++t) acc[mi][t] = (v8f)(0.f);

    FragB16 af[2][2], bfr[2][4];
#define LOAD_FRAGS(buf, kc)                                                    \
    {                                                                          \
        const unsigned short* pa = Abase + (kc) * 32;                          \
        af[buf][0].q[0] = *(const uint4*)(pa);                                 \
        af[buf][0].q[1] = *(const uint4*)(pa + 16);                            \
        af[buf][1].q[0] = *(const uint4*)(pa + 16 * lda);                      \
        af[buf][1].q[1] = *(const uint4*)(pa + 16 * lda + 16);                 \
        const unsigned short* pb = Bbase + (kc) * 512;                         \
        bfr[buf][0].q[0] = *(const uint4*)(pb);                                \
        bfr[buf][0].q[1] = *(const uint4*)(pb + 8);                            \
        bfr[buf][1].q[0] = *(const uint4*)(pb + (size_t)KCtot * 512);          \
        bfr[buf][1].q[1] = *(const uint4*)(pb + (size_t)KCtot * 512 + 8);      \
        bfr[buf][2].q[0] = *(const uint4*)(pb + (size_t)2 * KCtot * 512);      \
        bfr[buf][2].q[1] = *(const uint4*)(pb + (size_t)2 * KCtot * 512 + 8);  \
        bfr[buf][3].q[0] = *(const uint4*)(pb + (size_t)3 * KCtot * 512);      \
        bfr[buf][3].q[1] = *(const uint4*)(pb + (size_t)3 * KCtot * 512 + 8);  \
    }

    LOAD_FRAGS(0, 0);
#pragma unroll
    for (int kc = 0; kc < 16; ++kc) {
        const int cur = kc & 1, nxt = cur ^ 1;
        if (kc < 15) LOAD_FRAGS(nxt, kc + 1);
#pragma unroll
        for (int t = 0; t < 4; ++t) {
            acc[0][t] = wmma_bf16(af[cur][0], bfr[cur][t], acc[0][t]);
            acc[1][t] = wmma_bf16(af[cur][1], bfr[cur][t], acc[1][t]);
        }
    }
#undef LOAD_FRAGS

#pragma unroll
    for (int mi = 0; mi < 2; ++mi)
#pragma unroll
        for (int t = 0; t < 4; ++t)
#pragma unroll
            for (int r = 0; r < 8; ++r) {
                int m = mt * 32 + mi * 16 + lh * 8 + r;
                int n = ng * 64 + t * 16 + (lane & 15);
                float val = acc[mi][t][r];
                if (bias) val += bias[n];
                size_t o = (size_t)m * ldc + n;
                if (accFlag) val += C[o];
                C[o] = val;
                if (Cbf) Cbf[o] = f2bf(val);
            }
}

// ------ classifier GEMM: logits = [out3|attn] @ cls_W + b ------------------
// Block = 8 waves = 256 M-rows x 64 N-cols. B tiles staged to LDS with
// global_load_async_to_lds_b128 (double-buffered, ASYNCcnt-synchronized);
// every wave in the block shares the same B fragments (8x less L2 traffic).
__global__ __launch_bounds__(256) void gemm_cls_kernel(
    const unsigned short* __restrict__ A0,  // 2048x512 bf16 (normed out)
    const unsigned short* __restrict__ A1,  // 2048x512 bf16 (attn)
    const unsigned short* __restrict__ Bp,  // packed cls_W, KCtot = 32
    float* __restrict__ C, const float* __restrict__ bias) {
    __shared__ __align__(16) unsigned short sB[2][4 * 512];
    const int lane = threadIdx.x & 31;
    const int wave = threadIdx.x >> 5;
    const int lh   = lane >> 4;
    const int ng   = blockIdx.x % NGRP_CLS;
    const int mblk = blockIdx.x / NGRP_CLS;
    const int KCtot = 32;

    // async staging: each thread copies 16B of the 4KB B block for chunk kk
    const int ti  = threadIdx.x >> 6;   // tile 0..3
    const int sub = threadIdx.x & 63;   // 64 x 8 ushorts
    const unsigned short* SrcBase =
        Bp + (size_t)(ng * 4 + ti) * KCtot * 512 + sub * 8;
#define STAGE_ASYNC(buf, kk)                                                   \
    {                                                                          \
        unsigned ldsOff = (unsigned)(size_t)(&sB[buf][ti * 512 + sub * 8]);    \
        const void* gsrc = (const void*)(SrcBase + (size_t)(kk) * 512);        \
        asm volatile("global_load_async_to_lds_b128 %0, %1, off"               \
                     :: "v"(ldsOff), "v"(gsrc) : "memory");                    \
    }

    const size_t arow = (size_t)(mblk * 256 + wave * 32 + (lane & 15)) * 512 + lh * 8;
    const unsigned short* Ab0 = A0 + arow;
    const unsigned short* Ab1 = A1 + arow;

    v8f acc[2][4];
#pragma unroll
    for (int mi = 0; mi < 2; ++mi)
#pragma unroll
        for (int t = 0; t < 4; ++t) acc[mi][t] = (v8f)(0.f);

    FragB16 af[2][2];
#define LOAD_A(buf, seg, kc)                                                   \
    {                                                                          \
        const unsigned short* pa = ((seg) ? Ab1 : Ab0) + (kc) * 32;            \
        af[buf][0].q[0] = *(const uint4*)(pa);                                 \
        af[buf][0].q[1] = *(const uint4*)(pa + 16);                            \
        af[buf][1].q[0] = *(const uint4*)(pa + 16 * 512);                      \
        af[buf][1].q[1] = *(const uint4*)(pa + 16 * 512 + 16);                 \
    }

    STAGE_ASYNC(0, 0);
    LOAD_A(0, 0, 0);
    asm volatile("s_wait_asynccnt 0x0" ::: "memory");
    __syncthreads();

#pragma unroll
    for (int kk = 0; kk < 32; ++kk) {
        const int cur = kk & 1, nxt = cur ^ 1;
        if (kk < 31) {
            STAGE_ASYNC(nxt, kk + 1);
            LOAD_A(nxt, (kk + 1) >> 4, (kk + 1) & 15);
        }
        FragB16 bfr[4];
#pragma unroll
        for (int t = 0; t < 4; ++t) {
            bfr[t].q[0] = *(const uint4*)&sB[cur][t * 512 + lane * 16];
            bfr[t].q[1] = *(const uint4*)&sB[cur][t * 512 + lane * 16 + 8];
        }
#pragma unroll
        for (int t = 0; t < 4; ++t) {
            acc[0][t] = wmma_bf16(af[cur][0], bfr[t], acc[0][t]);
            acc[1][t] = wmma_bf16(af[cur][1], bfr[t], acc[1][t]);
        }
        if (kk < 31) {
            asm volatile("s_wait_asynccnt 0x0" ::: "memory");
            __syncthreads();
        }
    }
#undef STAGE_ASYNC
#undef LOAD_A

#pragma unroll
    for (int mi = 0; mi < 2; ++mi)
#pragma unroll
        for (int t = 0; t < 4; ++t)
#pragma unroll
            for (int r = 0; r < 8; ++r) {
                int m = mblk * 256 + wave * 32 + mi * 16 + lh * 8 + r;
                int n = ng * 64 + t * 16 + (lane & 15);
                C[(size_t)m * NWD_ + n] = acc[mi][t][r] + bias[n];
            }
}

// ------------- persistent LN-LSTM scan: 32 blocks x 256 threads -------------
__global__ __launch_bounds__(256) void lstm_scan_kernel(
    const float* __restrict__ Zx,
    const unsigned short* __restrict__ WhP, int kcOff, int KCtot,
    const float* __restrict__ g, const float* __restrict__ beta,
    const float* __restrict__ h0, const float* __restrict__ c0,
    const float* __restrict__ res,
    float* __restrict__ out, unsigned short* __restrict__ outbf,
    float* __restrict__ cbuf, unsigned short* __restrict__ hbf,
    float* __restrict__ zbuf, unsigned* __restrict__ bar) {
    const int lane  = threadIdx.x & 31;
    const int lh    = lane >> 4;
    const int gwave = blockIdx.x * 8 + (threadIdx.x >> 5);
    const int mt = gwave >> 7, nt = gwave & 127;
    const int b  = blockIdx.x;
    __shared__ float red[256], red2[256];

    for (int d = threadIdx.x; d < D_; d += 256) {
        cbuf[b * D_ + d] = c0[d];
        hbf[b * D_ + d]  = f2bf(h0[d]);
    }
    grid_barrier(bar, bar + 1, 32);

    const unsigned short* Abase = hbf + (size_t)(mt * 16 + (lane & 15)) * D_ + lh * 8;
    const unsigned short* Bbase = WhP + ((size_t)(nt * KCtot + kcOff) * 32 + lane) * 16;
    const int n = nt * 16 + (lane & 15);

    for (int t = 0; t < LT_; ++t) {
        v8f acc = (v8f)(0.f);
        FragB16 a2[2], b2[2];
#define LOAD_HW(buf, kc)                                           \
        {                                                          \
            const unsigned short* pa = Abase + (kc) * 32;          \
            a2[buf].q[0] = *(const uint4*)(pa);                    \
            a2[buf].q[1] = *(const uint4*)(pa + 16);               \
            const unsigned short* pb = Bbase + (kc) * 512;         \
            b2[buf].q[0] = *(const uint4*)(pb);                    \
            b2[buf].q[1] = *(const uint4*)(pb + 8);                \
        }
        LOAD_HW(0, 0);
#pragma unroll
        for (int kc = 0; kc < 16; ++kc) {
            const int cur = kc & 1, nxt = cur ^ 1;
            if (kc < 15) LOAD_HW(nxt, kc + 1);
            acc = wmma_bf16(a2[cur], b2[cur], acc);
        }
#undef LOAD_HW
#pragma unroll
        for (int r = 0; r < 8; ++r) {
            int mb = mt * 16 + lh * 8 + r;
            zbuf[(size_t)mb * 2048 + n] =
                acc[r] + Zx[((size_t)mb * LT_ + t) * 2048 + n];
        }
        grid_barrier(bar, bar + 1, 32);

        const float* zr = zbuf + (size_t)b * 2048;
        float s = 0.f, s2 = 0.f;
        for (int i = threadIdx.x; i < 2048; i += 256) {
            float v = zr[i]; s += v; s2 += v * v;
        }
        red[threadIdx.x] = s; red2[threadIdx.x] = s2;
        __syncthreads();
        for (int o = 128; o; o >>= 1) {
            if (threadIdx.x < o) {
                red[threadIdx.x]  += red[threadIdx.x + o];
                red2[threadIdx.x] += red2[threadIdx.x + o];
            }
            __syncthreads();
        }
        float mean = red[0] * (1.f / 2048.f);
        float var  = red2[0] * (1.f / 2048.f) - mean * mean;
        float rstd = rsqrtf(var + 1e-6f);
        for (int d = threadIdx.x; d < D_; d += 256) {
            float zi = (zr[d]        - mean) * rstd * g[d]        + beta[d];
            float zf = (zr[512 + d]  - mean) * rstd * g[512 + d]  + beta[512 + d];
            float zo = (zr[1024 + d] - mean) * rstd * g[1024 + d] + beta[1024 + d];
            float zg = (zr[1536 + d] - mean) * rstd * g[1536 + d] + beta[1536 + d];
            float ig = sigm(zi), fg = sigm(zf), og = sigm(zo), hid = tanhf(zg);
            float c = fg * cbuf[b * D_ + d] + ig * hid;
            float h = og * c;
            cbuf[b * D_ + d] = c;
            hbf[b * D_ + d]  = f2bf(h);
            float o2 = h;
            size_t oidx = ((size_t)b * LT_ + t) * D_ + d;
            if (res) o2 += res[oidx];
            out[oidx] = o2;
            if (outbf) outbf[oidx] = f2bf(o2);
        }
        grid_barrier(bar, bar + 1, 32);
    }
}

// -------- fused cross-attention per (b,h): QK^T, softmax, PV via WMMA -------
__global__ __launch_bounds__(256) void attn_kernel(
    const float* __restrict__ Qb,
    const float* __restrict__ KVb,
    const unsigned char* __restrict__ mask,
    unsigned short* __restrict__ attnObf) {
    __shared__ __align__(16) unsigned short sQ[64 * 64];
    __shared__ __align__(16) unsigned short sK[64 * 64];
    __shared__ __align__(16) unsigned short sV[64 * 64];
    __shared__ __align__(16) unsigned short sP[64 * 64];
    __shared__ float sS[64 * 64];
    const int b = blockIdx.x >> 3, h = blockIdx.x & 7;
    const int lane = threadIdx.x & 31, lh = lane >> 4;
    const int wave = threadIdx.x >> 5;

    for (int i = threadIdx.x; i < 64 * 64; i += 256) {
        int r = i >> 6, d = i & 63;
        sQ[i] = f2bf(Qb[(size_t)(b * 64 + r) * 512 + h * 64 + d]);
        sK[i] = f2bf(KVb[(size_t)(b * 64 + r) * 1024 + h * 64 + d]);
        sV[i] = f2bf(KVb[(size_t)(b * 64 + r) * 1024 + 512 + h * 64 + d]);
    }
    __syncthreads();

    for (int ti = 0; ti < 2; ++ti) {
        int tile = wave * 2 + ti;
        int mt = tile >> 2, nt = tile & 3;
        v8f acc = (v8f)(0.f);
        int m = mt * 16 + (lane & 15);
        int key = nt * 16 + (lane & 15);
#pragma unroll
        for (int kc = 0; kc < 2; ++kc) {
            FragB16 a, bb;
#pragma unroll
            for (int v = 0; v < 8; ++v) {
                int kk = kc * 32 + lh * 8 + ((v & 4) << 2) + ((v & 3) << 1);
                a.u[v] = *(const unsigned*)&sQ[m * 64 + kk];
                int kb = kc * 32 + lh * 8 + ((v & 4) << 2) + ((v & 3) << 1);
                bb.u[v] = *(const unsigned*)&sK[key * 64 + kb];
            }
            acc = wmma_bf16(a, bb, acc);
        }
#pragma unroll
        for (int r = 0; r < 8; ++r) {
            int mm = mt * 16 + lh * 8 + r;
            int nn = nt * 16 + (lane & 15);
            float sv = acc[r] * 0.125f;
            if (mask[b * LS_ + nn]) sv = -1e30f;
            sS[mm * 64 + nn] = sv;
        }
    }
    __syncthreads();

    if (threadIdx.x < 64) {
        int q = threadIdx.x;
        float mx = -1e30f;
        for (int k = 0; k < 64; ++k) mx = fmaxf(mx, sS[q * 64 + k]);
        float sum = 0.f;
        for (int k = 0; k < 64; ++k) sum += __expf(sS[q * 64 + k] - mx);
        float inv = 1.f / sum;
        for (int k = 0; k < 64; ++k)
            sP[q * 64 + k] = f2bf(__expf(sS[q * 64 + k] - mx) * inv);
    }
    __syncthreads();

    for (int ti = 0; ti < 2; ++ti) {
        int tile = wave * 2 + ti;
        int mt = tile >> 2, nt = tile & 3;
        v8f acc = (v8f)(0.f);
        int m = mt * 16 + (lane & 15);
        int n = nt * 16 + (lane & 15);
#pragma unroll
        for (int kc = 0; kc < 2; ++kc) {
            FragB16 a, bb;
#pragma unroll
            for (int v = 0; v < 8; ++v) {
                int kk = kc * 32 + lh * 8 + ((v & 4) << 2) + ((v & 3) << 1);
                a.u[v] = *(const unsigned*)&sP[m * 64 + kk];
                int k0 = kc * 32 + lh * 8 + ((v & 4) << 2) + ((v & 3) << 1);
                bb.u[v] = (unsigned)sV[k0 * 64 + n] | ((unsigned)sV[(k0 + 1) * 64 + n] << 16);
            }
            acc = wmma_bf16(a, bb, acc);
        }
#pragma unroll
        for (int r = 0; r < 8; ++r) {
            int mm = mt * 16 + lh * 8 + r;
            int nn = nt * 16 + (lane & 15);
            attnObf[(size_t)(b * 64 + mm) * 512 + h * 64 + nn] = f2bf(acc[r]);
        }
    }
}

// ------------------ final LN (per row over D, bf16 out) ---------------------
__global__ __launch_bounds__(256) void ln_rows_kernel(
    const float* __restrict__ X, const float* __restrict__ g,
    const float* __restrict__ bta, unsigned short* __restrict__ Ybf) {
    __shared__ float red[256], red2[256];
    int r = blockIdx.x;
    const float* xr = X + (size_t)r * D_;
    float s = 0.f, s2 = 0.f;
    for (int i = threadIdx.x; i < D_; i += 256) {
        float v = xr[i]; s += v; s2 += v * v;
    }
    red[threadIdx.x] = s; red2[threadIdx.x] = s2;
    __syncthreads();
    for (int o = 128; o; o >>= 1) {
        if (threadIdx.x < o) { red[threadIdx.x] += red[threadIdx.x + o]; red2[threadIdx.x] += red2[threadIdx.x + o]; }
        __syncthreads();
    }
    float mean = red[0] * (1.f / D_);
    float var  = red2[0] * (1.f / D_) - mean * mean;
    float rstd = rsqrtf(var + 1e-6f);
    for (int i = threadIdx.x; i < D_; i += 256)
        Ybf[(size_t)r * D_ + i] = f2bf((xr[i] - mean) * rstd * g[i] + bta[i]);
}

// --------------------------- in-place log-softmax ---------------------------
__global__ __launch_bounds__(256) void logsoftmax_kernel(float* __restrict__ X, int N) {
    __shared__ float red[256];
    int r = blockIdx.x;
    float* xr = X + (size_t)r * N;
    float mx = -1e30f;
    for (int i = threadIdx.x; i < N; i += 256) mx = fmaxf(mx, xr[i]);
    red[threadIdx.x] = mx; __syncthreads();
    for (int o = 128; o; o >>= 1) { if (threadIdx.x < o) red[threadIdx.x] = fmaxf(red[threadIdx.x], red[threadIdx.x + o]); __syncthreads(); }
    mx = red[0]; __syncthreads();
    float s = 0.f;
    for (int i = threadIdx.x; i < N; i += 256) s += __expf(xr[i] - mx);
    red[threadIdx.x] = s; __syncthreads();
    for (int o = 128; o; o >>= 1) { if (threadIdx.x < o) red[threadIdx.x] += red[threadIdx.x + o]; __syncthreads(); }
    float lse = mx + __logf(red[0]);
    for (int i = threadIdx.x; i < N; i += 256) xr[i] = xr[i] - lse;
}

// ============================== host launcher ===============================
extern "C" void kernel_launch(void* const* d_in, const int* in_sizes, int n_in,
                              void* d_out, int out_size, void* d_ws, size_t ws_size,
                              hipStream_t stream) {
    const float* inpute  = (const float*)d_in[0];
    const int*   inputo  = (const int*)d_in[1];
    const unsigned char* mask = (const unsigned char*)d_in[2];
    const float* wemb    = (const float*)d_in[3];
    const float* fl_W    = (const float*)d_in[4];
    const float* fl_b    = (const float*)d_in[5];
    const float* fl_g    = (const float*)d_in[6];
    const float* fl_beta = (const float*)d_in[7];
    const float* fl_h0   = (const float*)d_in[8];
    const float* fl_c0   = (const float*)d_in[9];
    const float* proj_W  = (const float*)d_in[10];
    const float* Wq      = (const float*)d_in[11];
    const float* bq      = (const float*)d_in[12];
    const float* Wkv     = (const float*)d_in[13];
    const float* bkv     = (const float*)d_in[14];
    const float* Wo      = (const float*)d_in[15];
    const float* bo      = (const float*)d_in[16];
    const float* ld_W    = (const float*)d_in[17];
    const float* ld_b    = (const float*)d_in[18];
    const float* ld_g    = (const float*)d_in[19];
    const float* ld_beta = (const float*)d_in[20];
    const float* ld_h0   = (const float*)d_in[21];
    const float* ld_c0   = (const float*)d_in[22];
    const float* norm_g  = (const float*)d_in[23];
    const float* norm_b  = (const float*)d_in[24];
    const float* cls_W   = (const float*)d_in[25];
    const float* cls_b   = (const float*)d_in[26];
    float* out_logits = (float*)d_out;

    char* ws = (char*)d_ws;
    size_t off = 0;
    auto alloc = [&](size_t bytes) -> char* {
        char* p = ws + off;
        off = (off + bytes + 255) & ~(size_t)255;
        return p;
    };
    unsigned*       bar   = (unsigned*)alloc(256);
    unsigned short* pFl   = (unsigned short*)alloc((size_t)1024 * 2048 * 2);
    unsigned short* pLd0  = (unsigned short*)alloc((size_t)1536 * 2048 * 2);
    unsigned short* pLd1  = (unsigned short*)alloc((size_t)1536 * 2048 * 2);
    unsigned short* pProj = (unsigned short*)alloc((size_t)512 * 512 * 2);
    unsigned short* pWq   = (unsigned short*)alloc((size_t)512 * 512 * 2);
    unsigned short* pWkv  = (unsigned short*)alloc((size_t)512 * 1024 * 2);
    unsigned short* pWo   = (unsigned short*)alloc((size_t)512 * 512 * 2);
    unsigned short* pCls  = (unsigned short*)alloc((size_t)1024 * 32000 * 2);
    float* Zx    = (float*)alloc((size_t)2048 * 2048 * 4);
    float* out0  = (float*)alloc((size_t)2048 * 512 * 4);
    float* out1  = (float*)alloc((size_t)2048 * 512 * 4);
    float* out2  = (float*)alloc((size_t)2048 * 512 * 4);
    float* enc   = (float*)alloc((size_t)2048 * 512 * 4);
    float* Qb    = (float*)alloc((size_t)2048 * 512 * 4);
    float* KVb   = (float*)alloc((size_t)2048 * 1024 * 4);
    float* attnB = (float*)alloc((size_t)2048 * 512 * 4);
    unsigned short* x0bf    = (unsigned short*)alloc((size_t)2048 * 512 * 2);
    unsigned short* inpbf   = (unsigned short*)alloc((size_t)2048 * 512 * 2);
    unsigned short* out0bf  = (unsigned short*)alloc((size_t)2048 * 512 * 2);
    unsigned short* out1bf  = (unsigned short*)alloc((size_t)2048 * 512 * 2);
    unsigned short* out3bf  = (unsigned short*)alloc((size_t)2048 * 512 * 2);
    unsigned short* encbf   = (unsigned short*)alloc((size_t)2048 * 512 * 2);
    unsigned short* attnObf = (unsigned short*)alloc((size_t)2048 * 512 * 2);
    unsigned short* attnBbf = (unsigned short*)alloc((size_t)2048 * 512 * 2);
    float* cbuf  = (float*)alloc((size_t)32 * 512 * 4);
    unsigned short* hbf = (unsigned short*)alloc((size_t)32 * 512 * 2);
    float* zbuf  = (float*)alloc((size_t)32 * 2048 * 4);

    init_bar_kernel<<<1, 64, 0, stream>>>(bar);

    auto pack = [&](const float* W, unsigned short* P, int K, int N) {
        int total = (K >> 5) * (N >> 4) * 32;
        pack_w_kernel<<<(total + 255) / 256, 256, 0, stream>>>(W, P, K, N);
    };
    pack(fl_W, pFl, 1024, 2048);
    pack(ld_W, pLd0, 1536, 2048);
    pack(ld_W + (size_t)1536 * 2048, pLd1, 1536, 2048);
    pack(proj_W, pProj, 512, 512);
    pack(Wq, pWq, 512, 512);
    pack(Wkv, pWkv, 512, 1024);
    pack(Wo, pWo, 512, 512);
    pack(cls_W, pCls, 1024, 32000);

    embed_kernel<<<(B_ * LT_ * D_ + 255) / 256, 256, 0, stream>>>(inputo, wemb, x0bf);
    cvt_bf16_kernel<<<(2048 * 512 + 255) / 256, 256, 0, stream>>>(inpute, inpbf, 2048 * 512);

    auto gemm = [&](const unsigned short* A, const unsigned short* Bp, int kcOff, int KCtot,
                    float* C, int ldc, const float* bias, unsigned short* Cbf,
                    int M, int N, int acc) {
        int waves = (M >> 5) * (N >> 6);
        gemm_ws_kernel<<<(waves + 7) / 8, 256, 0, stream>>>(A, 512, Bp, kcOff, KCtot,
                                                            C, ldc, bias, Cbf, M, N, acc);
    };

    // ---- encoder-side LSTM ----
    gemm(x0bf, pFl, 0, 32, Zx, 2048, fl_b, nullptr, 2048, 2048, 0);
    lstm_scan_kernel<<<32, 256, 0, stream>>>(Zx, pFl, 16, 32, fl_g, fl_beta,
                                             fl_h0, fl_c0, nullptr, out0, out0bf,
                                             cbuf, hbf, zbuf, bar);

    // ---- cross attention ----
    gemm(inpbf, pProj, 0, 16, enc, 512, nullptr, encbf, 2048, 512, 0);
    gemm(out0bf, pWq, 0, 16, Qb, 512, bq, nullptr, 2048, 512, 0);
    gemm(encbf, pWkv, 0, 16, KVb, 1024, bkv, nullptr, 2048, 1024, 0);
    attn_kernel<<<B_ * NH_, 256, 0, stream>>>(Qb, KVb, mask, attnObf);
    gemm(attnObf, pWo, 0, 16, attnB, 512, bo, attnBbf, 2048, 512, 0);

    // ---- decoder LSTM layer 0 ----
    gemm(out0bf, pLd0, 0, 48, Zx, 2048, ld_b, nullptr, 2048, 2048, 0);
    gemm(attnBbf, pLd0, 16, 48, Zx, 2048, nullptr, nullptr, 2048, 2048, 1);
    lstm_scan_kernel<<<32, 256, 0, stream>>>(Zx, pLd0, 32, 48, ld_g, ld_beta,
                                             ld_h0, ld_c0, nullptr, out1, out1bf,
                                             cbuf, hbf, zbuf, bar);

    // ---- decoder LSTM layer 1 (residual) ----
    gemm(out1bf, pLd1, 0, 48, Zx, 2048, ld_b + 2048, nullptr, 2048, 2048, 0);
    gemm(attnBbf, pLd1, 16, 48, Zx, 2048, nullptr, nullptr, 2048, 2048, 1);
    lstm_scan_kernel<<<32, 256, 0, stream>>>(Zx, pLd1, 32, 48, ld_g + 2048, ld_beta + 2048,
                                             ld_h0 + 512, ld_c0 + 512, out1, out2, nullptr,
                                             cbuf, hbf, zbuf, bar);

    // ---- final LN + fused concat classifier (async-LDS staged B) ----
    ln_rows_kernel<<<2048, 256, 0, stream>>>(out2, norm_g, norm_b, out3bf);
    gemm_cls_kernel<<<8 * NGRP_CLS, 256, 0, stream>>>(out3bf, attnBbf, pCls,
                                                      out_logits, cls_b);
    logsoftmax_kernel<<<2048, 256, 0, stream>>>(out_logits, NWD_);
}
